// MultiStreamSparseAutoencoder_83940840833381
// MI455X (gfx1250) — matrix operand order repaired and checked by reading, not compiled, not run on hardware
//
#include <hip/hip_runtime.h>
#include <hip/hip_bf16.h>
#include <math.h>

typedef __bf16 bf16_t;
typedef __attribute__((ext_vector_type(16))) __bf16 v16bf;
typedef __attribute__((ext_vector_type(8)))  __bf16 v8bf;
typedef __attribute__((ext_vector_type(8)))  float  v8f;

#define B_    2048
#define L_    16384
#define DV_   768
#define DT_   1024
#define K_    64
#define AUXK_ 256

// ---------------------------------------------------------------- helpers
__device__ __forceinline__ bf16_t f2bf(float f) {
  unsigned u = __builtin_bit_cast(unsigned, f);
  unsigned r = u + 0x7FFFu + ((u >> 16) & 1u);   // round-to-nearest-even
  unsigned short h = (unsigned short)(r >> 16);
  return __builtin_bit_cast(bf16_t, h);
}

// xc = (x - pre_b) cast to bf16
__global__ void sub_bias_bf16_kernel(const float* __restrict__ x,
                                     const float* __restrict__ pb,
                                     bf16_t* __restrict__ out, int D, int total) {
  int i = blockIdx.x * blockDim.x + threadIdx.x;
  if (i < total) out[i] = f2bf(x[i] - pb[i % D]);
}

__global__ void cast_bf16_kernel(const float* __restrict__ in,
                                 bf16_t* __restrict__ out, int n) {
  int i = blockIdx.x * blockDim.x + threadIdx.x;
  if (i < n) out[i] = f2bf(in[i]);
}

// ------------------------------------------------- encoder GEMM (bf16 WMMA)
// lg[b,l] = sum_d Xc[b,d]*W[l,d] + lat_b[l]
// One wave computes a 16(M) x 128(N) C-strip: 8 accumulators sharing the A frag.
// A fragment (16x32 bf16, ISA layout): lane m=lane&15, hi=lane>>4 holds
//   K = hi*8 + 0..7  and  K = 16 + hi*8 + 0..7  (two contiguous 16B chunks).
// B fragment (32x16 bf16): lane col n=lane&15, hi holds
//   K = hi*8 + 0..7  and  K = 16 + hi*8 + 0..7 of W row (l, :) (same as A).
// D is a template parameter so all tile offsets become 24-bit immediates.
template <int D>
__global__ __launch_bounds__(256) void wmma_encode_kernel(
    const bf16_t* __restrict__ X, const bf16_t* __restrict__ W,
    const float* __restrict__ lat_b, float* __restrict__ lg) {
  const int lane = threadIdx.x & 31;
  const int wave = threadIdx.x >> 5;
  const int hi   = lane >> 4;
  const int lr   = lane & 15;
  const int r0   = blockIdx.y << 4;                    // M tile base (batch)
  const int n0   = (blockIdx.x * 8 + wave) << 7;       // N strip base (latent)

  v8f acc[8] = {};

  const bf16_t* xrow  = X + (size_t)(r0 + lr) * D + hi * 8;
  const bf16_t* wbase = W + (size_t)(n0 + lr) * D + hi * 8;
  for (int k0 = 0; k0 < D; k0 += 32) {
    v8bf alo = *(const v8bf*)(xrow + k0);
    v8bf ahi = *(const v8bf*)(xrow + k0 + 16);
    v16bf a = __builtin_shufflevector(alo, ahi,
        0,1,2,3,4,5,6,7,8,9,10,11,12,13,14,15);
#pragma unroll
    for (int t = 0; t < 8; ++t) {
      const bf16_t* wp = wbase + (size_t)t * 16 * D + k0;
      __builtin_prefetch(wp + 32, 0, 1);   // unconditional global_prefetch_b8
      v8bf b0 = *(const v8bf*)(wp);
      v8bf b1 = *(const v8bf*)(wp + 16);
      v16bf bb = __builtin_shufflevector(b0, b1,
          0,1,2,3,4,5,6,7,8,9,10,11,12,13,14,15);
      acc[t] = __builtin_amdgcn_wmma_f32_16x16x32_bf16(
          /*neg_a=*/false, a, /*neg_b=*/false, bb,
          /*c_mod=*/(short)0, acc[t], /*reuse_a=*/false, /*reuse_b=*/false);
    }
  }
  // C layout: VGPR v, lanes 0-15 -> (M=v, N=lane); lanes 16-31 -> (M=8+v, N=lane-16)
#pragma unroll
  for (int t = 0; t < 8; ++t) {
    const int col = n0 + t * 16 + lr;
    const float lb = lat_b[col];
    float* outp = lg + (size_t)(r0 + hi * 8) * L_ + col;
#pragma unroll
    for (int v = 0; v < 8; ++v)
      outp[(size_t)v * L_] = acc[t][v] + lb;
  }
}

// ------------------------------------------ shared global Top-K (K=64) gate
__global__ __launch_bounds__(256) void topk_shared_kernel(
    const float* __restrict__ lg_v, const float* __restrict__ lg_t,
    int* __restrict__ sidx, float* __restrict__ actv, float* __restrict__ actt,
    float* __restrict__ codes_v, float* __restrict__ codes_t,
    float* __restrict__ sidx_f,
    unsigned* __restrict__ flags_v, unsigned* __restrict__ flags_t) {
  extern __shared__ float s_val[];   // L_ floats (64 KB of the 320 KB WGP LDS)
  __shared__ float s_rv[256];
  __shared__ int   s_ri[256];
  const int b = blockIdx.x, tid = threadIdx.x;
  const float* pv = lg_v + (size_t)b * L_;
  const float* pt = lg_t + (size_t)b * L_;
  for (int i = tid; i < L_; i += 256) s_val[i] = pv[i] + pt[i];
  __syncthreads();
  for (int k = 0; k < K_; ++k) {
    float bv = -INFINITY; int bi = L_;
    for (int i = tid; i < L_; i += 256) {
      float v = s_val[i];
      if (v > bv) { bv = v; bi = i; }      // strict > keeps lowest index on ties
    }
    s_rv[tid] = bv; s_ri[tid] = bi;
    __syncthreads();
    for (int s = 128; s > 0; s >>= 1) {
      if (tid < s) {
        float ov = s_rv[tid + s]; int oi = s_ri[tid + s];
        if (ov > s_rv[tid] || (ov == s_rv[tid] && oi < s_ri[tid])) {
          s_rv[tid] = ov; s_ri[tid] = oi;
        }
      }
      __syncthreads();
    }
    const int idx = s_ri[0];
    if (tid == 0) {
      float vv = pv[idx], vt = pt[idx];
      float av = vv > 0.f ? vv : 0.f;
      float at = vt > 0.f ? vt : 0.f;
      sidx[b * K_ + k]   = idx;
      actv[b * K_ + k]   = av;
      actt[b * K_ + k]   = at;
      sidx_f[b * K_ + k] = (float)idx;
      codes_v[(size_t)b * L_ + idx] = av;
      codes_t[(size_t)b * L_ + idx] = at;
      if (av > 1e-3f) flags_v[idx] = 1u;   // racy all-write-1: benign
      if (at > 1e-3f) flags_t[idx] = 1u;
      s_val[idx] = -INFINITY;
    }
    __syncthreads();
  }
}

// new_stats = active ? 1 : stats+1 ; dead = new_stats > 1000
__global__ void stats_dead_kernel(const unsigned* __restrict__ flags,
                                  const int* __restrict__ stats,
                                  unsigned* __restrict__ dead) {
  int l = blockIdx.x * blockDim.x + threadIdx.x;
  if (l < L_) {
    int ns = flags[l] ? 1 : stats[l] + 1;
    dead[l] = (ns > 1000) ? 1u : 0u;
  }
}

// -------------------------------------------- AuxK per-stream top-256 + scatter
__global__ __launch_bounds__(256) void auxk_kernel(
    const float* __restrict__ lg, const unsigned* __restrict__ dead,
    float* __restrict__ acodes) {
  extern __shared__ float s_val[];
  __shared__ float s_rv[256];
  __shared__ int   s_ri[256];
  const int b = blockIdx.x, tid = threadIdx.x;
  const float* p = lg + (size_t)b * L_;
  for (int i = tid; i < L_; i += 256) s_val[i] = dead[i] ? p[i] : 0.0f;
  __syncthreads();
  for (int k = 0; k < AUXK_; ++k) {
    float bv = -INFINITY; int bi = L_;
    for (int i = tid; i < L_; i += 256) {
      float v = s_val[i];
      if (v > bv) { bv = v; bi = i; }
    }
    s_rv[tid] = bv; s_ri[tid] = bi;
    __syncthreads();
    for (int s = 128; s > 0; s >>= 1) {
      if (tid < s) {
        float ov = s_rv[tid + s]; int oi = s_ri[tid + s];
        if (ov > s_rv[tid] || (ov == s_rv[tid] && oi < s_ri[tid])) {
          s_rv[tid] = ov; s_ri[tid] = oi;
        }
      }
      __syncthreads();
    }
    const int idx = s_ri[0];
    if (tid == 0) {
      float v = s_val[idx];
      acodes[(size_t)b * L_ + idx] = v > 0.f ? v : 0.f;
      s_val[idx] = -INFINITY;
    }
    __syncthreads();
  }
}

// ------------------------------------ sparse decode (Wdec L2-resident gathers)
__global__ __launch_bounds__(256) void decode_kernel(
    const int* __restrict__ sidx, const float* __restrict__ actv,
    const float* __restrict__ actt,
    const float* __restrict__ Wdec_v, const float* __restrict__ Wdec_t,
    const float* __restrict__ pre_b_v, const float* __restrict__ pre_b_t,
    float* __restrict__ recon_v, float* __restrict__ recon_t,
    float* __restrict__ cross_tv, float* __restrict__ cross_vt) {
  __shared__ int   s_idx[K_];
  __shared__ float s_av[K_], s_at[K_];
  const int b = blockIdx.x, tid = threadIdx.x;
  if (tid < K_) {
    s_idx[tid] = sidx[b * K_ + tid];
    s_av[tid]  = actv[b * K_ + tid];
    s_at[tid]  = actt[b * K_ + tid];
  }
  __syncthreads();
  for (int d = tid; d < DV_; d += 256) {
    const float* wrow = Wdec_v + (size_t)d * L_;
    float a0 = 0.f, a1 = 0.f;
#pragma unroll 8
    for (int k = 0; k < K_; ++k) {
      float w = wrow[s_idx[k]];
      a0 += s_av[k] * w;        // recon_v
      a1 += s_at[k] * w;        // cross_v_from_t (shares Wdec_v load)
    }
    recon_v [(size_t)b * DV_ + d] = a0 + pre_b_v[d];
    cross_vt[(size_t)b * DV_ + d] = a1 + pre_b_v[d];
  }
  for (int d = tid; d < DT_; d += 256) {
    const float* wrow = Wdec_t + (size_t)d * L_;
    float a0 = 0.f, a1 = 0.f;
#pragma unroll 8
    for (int k = 0; k < K_; ++k) {
      float w = wrow[s_idx[k]];
      a0 += s_at[k] * w;        // recon_t
      a1 += s_av[k] * w;        // cross_t_from_v
    }
    recon_t [(size_t)b * DT_ + d] = a0 + pre_b_t[d];
    cross_tv[(size_t)b * DT_ + d] = a1 + pre_b_t[d];
  }
}

__global__ void finalize_kernel(float* __restrict__ avg) {
  if (blockIdx.x == 0 && threadIdx.x == 0) *avg = 64.0f;  // exactly K per row/stream
}

// ---------------------------------------------------------------- launcher
extern "C" void kernel_launch(void* const* d_in, const int* in_sizes, int n_in,
                              void* d_out, int out_size, void* d_ws, size_t ws_size,
                              hipStream_t stream) {
  (void)in_sizes; (void)n_in; (void)out_size; (void)ws_size;

  const float* x_v     = (const float*)d_in[0];
  const float* x_t     = (const float*)d_in[1];
  const float* Wenc_v  = (const float*)d_in[2];
  const float* Wdec_v  = (const float*)d_in[3];
  const float* pre_b_v = (const float*)d_in[4];
  const float* lat_b_v = (const float*)d_in[5];
  const int*   stats_v = (const int*)d_in[6];
  const float* Wenc_t  = (const float*)d_in[7];
  const float* Wdec_t  = (const float*)d_in[8];
  const float* pre_b_t = (const float*)d_in[9];
  const float* lat_b_t = (const float*)d_in[10];
  const int*   stats_t = (const int*)d_in[11];
  float* out = (float*)d_out;

  // output layout (floats), reference tuple order
  constexpr size_t O_RV  = 0;
  constexpr size_t O_RT  = O_RV + (size_t)B_ * DV_;
  constexpr size_t O_CTV = O_RT + (size_t)B_ * DT_;
  constexpr size_t O_CVT = O_CTV + (size_t)B_ * DT_;
  constexpr size_t O_CV  = O_CVT + (size_t)B_ * DV_;
  constexpr size_t O_CT  = O_CV + (size_t)B_ * L_;
  constexpr size_t O_AV  = O_CT + (size_t)B_ * L_;
  constexpr size_t O_AT  = O_AV + (size_t)B_ * L_;
  constexpr size_t O_SI  = O_AT + (size_t)B_ * L_;
  constexpr size_t O_AVG = O_SI + (size_t)B_ * K_;

  // workspace carve (256B aligned)
  char* ws = (char*)d_ws;
  size_t off = 0;
  auto carve = [&](size_t bytes) -> void* {
    void* p = ws + off;
    off += (bytes + 255) & ~(size_t)255;
    return p;
  };
  float*  lg_v  = (float*)carve((size_t)B_ * L_ * 4);
  float*  lg_t  = (float*)carve((size_t)B_ * L_ * 4);
  bf16_t* wvb   = (bf16_t*)carve((size_t)L_ * DV_ * 2);
  bf16_t* wtb   = (bf16_t*)carve((size_t)L_ * DT_ * 2);
  bf16_t* xcv   = (bf16_t*)carve((size_t)B_ * DV_ * 2);
  bf16_t* xct   = (bf16_t*)carve((size_t)B_ * DT_ * 2);
  int*    sidx  = (int*)carve((size_t)B_ * K_ * 4);
  float*  actv  = (float*)carve((size_t)B_ * K_ * 4);
  float*  actt  = (float*)carve((size_t)B_ * K_ * 4);
  unsigned* flags_v = (unsigned*)carve((size_t)L_ * 4);
  unsigned* flags_t = (unsigned*)carve((size_t)L_ * 4);
  unsigned* dead_v  = (unsigned*)carve((size_t)L_ * 4);
  unsigned* dead_t  = (unsigned*)carve((size_t)L_ * 4);

  // 1) bf16 staging
  sub_bias_bf16_kernel<<<(B_ * DV_ + 255) / 256, 256, 0, stream>>>(x_v, pre_b_v, xcv, DV_, B_ * DV_);
  sub_bias_bf16_kernel<<<(B_ * DT_ + 255) / 256, 256, 0, stream>>>(x_t, pre_b_t, xct, DT_, B_ * DT_);
  cast_bf16_kernel<<<((size_t)L_ * DV_ + 255) / 256, 256, 0, stream>>>(Wenc_v, wvb, L_ * DV_);
  cast_bf16_kernel<<<((size_t)L_ * DT_ + 255) / 256, 256, 0, stream>>>(Wenc_t, wtb, L_ * DT_);

  // 2) WMMA encoder GEMMs: lg = Xc @ Wenc^T + lat_b
  dim3 eg(L_ / 1024, B_ / 16);   // 8 waves/block x 128 cols/wave = 1024 cols/block
  wmma_encode_kernel<DV_><<<eg, 256, 0, stream>>>(xcv, wvb, lat_b_v, lg_v);
  wmma_encode_kernel<DT_><<<eg, 256, 0, stream>>>(xct, wtb, lat_b_t, lg_t);

  // 3) zero sparse outputs (codes_v..acodes_t contiguous) and flags
  hipMemsetAsync(out + O_CV, 0, (size_t)4 * B_ * L_ * sizeof(float), stream);
  hipMemsetAsync(flags_v, 0, (size_t)2 * L_ * sizeof(unsigned), stream);

  // 4) shared Top-K gate + scatter codes + activity flags
  topk_shared_kernel<<<B_, 256, L_ * sizeof(float), stream>>>(
      lg_v, lg_t, sidx, actv, actt, out + O_CV, out + O_CT, out + O_SI,
      flags_v, flags_t);

  // 5) dead-neuron masks
  stats_dead_kernel<<<L_ / 256, 256, 0, stream>>>(flags_v, stats_v, dead_v);
  stats_dead_kernel<<<L_ / 256, 256, 0, stream>>>(flags_t, stats_t, dead_t);

  // 6) AuxK per-stream top-256 + scatter acodes
  auxk_kernel<<<B_, 256, L_ * sizeof(float), stream>>>(lg_v, dead_v, out + O_AV);
  auxk_kernel<<<B_, 256, L_ * sizeof(float), stream>>>(lg_t, dead_t, out + O_AT);

  // 7) sparse decode (4 outputs, Wdec rows gathered from L2)
  decode_kernel<<<B_, 256, 0, stream>>>(sidx, actv, actt, Wdec_v, Wdec_t,
                                        pre_b_v, pre_b_t,
                                        out + O_RV, out + O_RT,
                                        out + O_CTV, out + O_CVT);

  // 8) avg_active scalar
  finalize_kernel<<<1, 1, 0, stream>>>(out + O_AVG);
}
